// FastRCNNOutputLayers_MLN_27539330302078
// MI455X (gfx1250) — matrix operand order repaired and checked by reading, not compiled, not run on hardware
//
#include <hip/hip_runtime.h>
#include <hip/hip_bf16.h>

typedef __attribute__((ext_vector_type(16))) __bf16 v16bf;
typedef __attribute__((ext_vector_type(8)))  __bf16 v8bf;
typedef __attribute__((ext_vector_type(4)))  __bf16 v4bf;
typedef __attribute__((ext_vector_type(8)))  float  v8f;

#define DIMK   1024
#define MIX    5
#define CLS    81
#define BOXW   320
#define NTILES 73            // 1 (pi) + 26 (mu) + 26 (sigma) + 20 (box)
#define KCH    32            // 1024 / 32
#define ROWS   128           // rows per workgroup
#define WGT    128           // threads per workgroup (4 waves x 32 rows each)
#define XSTR   1032          // padded bf16 row stride in LDS
#define S1W    84            // padded width for s1[row][c]

// LDS byte offsets
#define OFF_XS   0
#define OFF_S1   (ROWS*XSTR*2)                 // 264192
#define OFF_PI   (OFF_S1 + ROWS*S1W*4)         // +43008
#define OFF_IDX  (OFF_PI + ROWS*8*4)           // +4096
#define OFF_T1   (OFF_IDX + ROWS*4)            // +512
#define OFF_AL   (OFF_T1 + ROWS*4)             // +512
#define SMEM_SZ  (OFF_AL + ROWS*4)             // 312832 bytes

// ---------------------------------------------------------------------------
// Pack the four fp32 weight matrices into one bf16 B-fragment-native buffer:
// Bp[((t*32 + kc)*32 + lane)*16 + e] = W[kc*32 + (lane>>4)*16 + e][tileCol(t, lane&15)]
// Lanes 0-15 carry K 0..15 of a 32-chunk for column n, lanes 16-31 carry K 16..31.
// Also packs biases: BiasP[t*16 + n].
// ---------------------------------------------------------------------------
__global__ __launch_bounds__(256) void mln_pack(
    const float* __restrict__ Wpi, const float* __restrict__ bpi,
    const float* __restrict__ Wmu, const float* __restrict__ bmu,
    const float* __restrict__ Wsg, const float* __restrict__ bsg,
    const float* __restrict__ Wbx, const float* __restrict__ bbx,
    __bf16* __restrict__ Bp, float* __restrict__ BiasP)
{
    const int tid  = blockIdx.x * 256 + threadIdx.x;
    const int pair = tid >> 5;          // (t, kc)
    const int lane = tid & 31;

    if (pair < NTILES * KCH) {
        const int t  = pair >> 5;
        const int kc = pair & 31;
        const int n  = lane & 15;
        const int kh = lane >> 4;

        const float* W; int ncols, col;
        if (t == 0)       { W = Wpi; ncols = MIX;  col = n; }
        else if (t < 27)  { W = Wmu; ncols = MIX*CLS; col = (t - 1)  * 16 + n; }
        else if (t < 53)  { W = Wsg; ncols = MIX*CLS; col = (t - 27) * 16 + n; }
        else              { W = Wbx; ncols = BOXW;    col = (t - 53) * 16 + n; }

        __bf16* dst = Bp + ((size_t)pair * 32 + lane) * 16;
        if (col < ncols) {
            const float* src = W + col;
            const int k0 = kc * 32 + kh * 16;
            #pragma unroll
            for (int e = 0; e < 16; ++e)
                dst[e] = (__bf16)src[(size_t)(k0 + e) * ncols];
        } else {
            #pragma unroll
            for (int e = 0; e < 16; ++e) dst[e] = (__bf16)0.0f;
        }
    }

    if (tid < NTILES * 16) {
        const int t = tid >> 4, n = tid & 15;
        float b = 0.0f;
        if (t == 0)      { if (n < MIX) b = bpi[n]; }
        else if (t < 27) { int c = (t - 1)  * 16 + n; if (c < MIX*CLS) b = bmu[c]; }
        else if (t < 53) { int c = (t - 27) * 16 + n; if (c < MIX*CLS) b = bsg[c]; }
        else             { int c = (t - 53) * 16 + n; if (c < BOXW)    b = bbx[c]; }
        BiasP[tid] = b;
    }
}

// ---------------------------------------------------------------------------
// Fused MLN head: bf16 WMMA GEMM over all 73 column tiles + full epilogue.
// 128 threads = 4 waves; wave w owns rows 32w..32w+31 (two 16-row A tiles).
// Register blocking is 2x2: per k-step, 2 A frags + 2 B frags feed 4 WMMAs
// (1 ds_load_b128 AND 1 global_load_b128 per WMMA -> LDS and WGP$ balanced).
// ---------------------------------------------------------------------------
__global__ __launch_bounds__(WGT) void mln_fused(
    const float* __restrict__ x, const __bf16* __restrict__ Bp,
    const float* __restrict__ BiasP, float* __restrict__ out, int N)
{
    extern __shared__ __align__(16) char smem[];
    __bf16* xs  = (__bf16*)(smem + OFF_XS);
    float*  s1  = (float*) (smem + OFF_S1);
    float*  lpi = (float*) (smem + OFF_PI);
    int*    lix = (int*)   (smem + OFF_IDX);
    float*  t1a = (float*) (smem + OFF_T1);
    float*  ala = (float*) (smem + OFF_AL);

    const int tid = threadIdx.x;
    const size_t rowBase = (size_t)blockIdx.x * ROWS;

    // ---- stage x tile: fp32 global -> bf16 LDS (coalesced float4 reads) ----
    const float4* xg = (const float4*)(x + rowBase * DIMK);
    for (int f = tid; f < ROWS * (DIMK / 4); f += WGT) {
        const int row = f >> 8;          // 256 float4 per row
        const int c4  = f & 255;
        float4 v = xg[f];
        v4bf b;
        b[0] = (__bf16)v.x; b[1] = (__bf16)v.y;
        b[2] = (__bf16)v.z; b[3] = (__bf16)v.w;
        *(v4bf*)(xs + (size_t)row * XSTR + c4 * 4) = b;
    }
    for (int i = tid; i < ROWS * S1W; i += WGT) s1[i] = 0.0f;
    t1a[tid] = 0.0f;            // WGT == ROWS: one row per thread
    ala[tid] = 0.0f;
    __syncthreads();

    const int lane = tid & 31;
    const int w    = tid >> 5;      // wave id: rows 32w..32w+31
    const int hi   = lane >> 4;     // lane half (selects K sub-range)
    const int n    = lane & 15;     // column within tile / row within A tile
    const int rb0  = 32 * w;        // first  row tile base
    const int rb1  = 32 * w + 16;   // second row tile base

    // A-fragment bases for this lane (contiguous K: lane half hi holds K 16*hi..16*hi+15)
    const __bf16* arow0 = xs + (size_t)(rb0 + n) * XSTR + hi * 16;
    const __bf16* arow1 = arow0 + (size_t)16 * XSTR;
    const v16bf* bpv = (const v16bf*)Bp;

    #define LOAD_A(base, kc)                                                  \
        __builtin_shufflevector(*(const v8bf*)((base) + (kc) * 32),           \
                                *(const v8bf*)((base) + (kc) * 32 + 8),       \
                                0,1,2,3,4,5,6,7,8,9,10,11,12,13,14,15)

    // pi tile: one column tile x two row tiles (B frag shared)
    auto gemm_pi = [&](v8f& o0, v8f& o1) {
        v8f a0 = {0.f,0.f,0.f,0.f,0.f,0.f,0.f,0.f};
        v8f a1 = {0.f,0.f,0.f,0.f,0.f,0.f,0.f,0.f};
        const v16bf* bt = bpv + lane;
        #pragma unroll 4
        for (int kc = 0; kc < KCH; ++kc) {
            v16bf af0 = LOAD_A(arow0, kc);
            v16bf af1 = LOAD_A(arow1, kc);
            v16bf b = bt[(size_t)kc * 32];
            a0 = __builtin_amdgcn_wmma_f32_16x16x32_bf16(
                     false, af0, false, b, (short)0, a0, false, false);
            a1 = __builtin_amdgcn_wmma_f32_16x16x32_bf16(
                     false, af1, false, b, (short)0, a1, false, false);
        }
        const float bias = BiasP[n];
        #pragma unroll
        for (int i = 0; i < 8; ++i) { a0[i] += bias; a1[i] += bias; }
        o0 = a0; o1 = a1;
    };

    // 2x2 block: tiles (t, t+1) x row tiles (rb0, rb1); 4 independent chains
    auto gemm_quad = [&](int t, v8f& o00, v8f& o01, v8f& o10, v8f& o11) {
        v8f a00 = {0.f,0.f,0.f,0.f,0.f,0.f,0.f,0.f};
        v8f a01 = a00, a10 = a00, a11 = a00;
        const v16bf* bt0 = bpv + (size_t)t * (KCH * 32) + lane;
        const v16bf* bt1 = bt0 + (KCH * 32);
        #pragma unroll 4
        for (int kc = 0; kc < KCH; ++kc) {
            v16bf af0 = LOAD_A(arow0, kc);
            v16bf af1 = LOAD_A(arow1, kc);
            v16bf b0 = bt0[(size_t)kc * 32];
            v16bf b1 = bt1[(size_t)kc * 32];
            a00 = __builtin_amdgcn_wmma_f32_16x16x32_bf16(
                      false, af0, false, b0, (short)0, a00, false, false);
            a01 = __builtin_amdgcn_wmma_f32_16x16x32_bf16(
                      false, af0, false, b1, (short)0, a01, false, false);
            a10 = __builtin_amdgcn_wmma_f32_16x16x32_bf16(
                      false, af1, false, b0, (short)0, a10, false, false);
            a11 = __builtin_amdgcn_wmma_f32_16x16x32_bf16(
                      false, af1, false, b1, (short)0, a11, false, false);
        }
        const float bias0 = BiasP[t * 16 + n];
        const float bias1 = BiasP[(t + 1) * 16 + n];
        #pragma unroll
        for (int i = 0; i < 8; ++i) {
            a00[i] += bias0; a01[i] += bias1;
            a10[i] += bias0; a11[i] += bias1;
        }
        o00 = a00; o01 = a01; o10 = a10; o11 = a11;
    };

    float* outScores = out;
    float* outDeltas = out + (size_t)N * CLS;
    float* outEpis   = out + (size_t)N * (CLS + BOXW);
    float* outAlea   = out + (size_t)N * (CLS + BOXW + 1);

    // -------- tile 0: pi logits -> softmax + argmax (staged through s1) -----
    {
        v8f p0, p1;
        gemm_pi(p0, p1);
        #pragma unroll
        for (int i = 0; i < 8; ++i) {
            s1[(size_t)(rb0 + i + 8 * hi) * S1W + n] = p0[i];
            s1[(size_t)(rb1 + i + 8 * hi) * S1W + n] = p1[i];
        }
        // all 32 lanes: one of the wave's 32 rows each
        const int row = rb0 + lane;
        float v[MIX];
        #pragma unroll
        for (int m = 0; m < MIX; ++m) v[m] = s1[(size_t)row * S1W + m];
        float mx = v[0]; int am = 0;
        #pragma unroll
        for (int m = 1; m < MIX; ++m) if (v[m] > mx) { mx = v[m]; am = m; }
        float e[MIX], sum = 0.0f;
        #pragma unroll
        for (int m = 0; m < MIX; ++m) { e[m] = __expf(v[m] - mx); sum += e[m]; }
        const float inv = 1.0f / sum;
        #pragma unroll
        for (int m = 0; m < MIX; ++m) lpi[row * 8 + m] = e[m] * inv;
        lix[row] = am;
        #pragma unroll
        for (int m = 0; m < 16; ++m) s1[(size_t)row * S1W + m] = 0.0f;
    }

    float t1r[16] = {0,0,0,0,0,0,0,0,0,0,0,0,0,0,0,0}; // sum pi*mu^2 parts
    float alr[16] = {0,0,0,0,0,0,0,0,0,0,0,0,0,0,0,0}; // sum pi*softplus parts

    // epilogue handlers for one finished tile accumulator -------------------
    auto mu_ep = [&](int t, int rb, int ib, v8f acc) {
        const int j = (t - 1) * 16 + n;
        if (j < MIX * CLS) {
            const int m = j / CLS;
            const int c = j - m * CLS;
            #pragma unroll
            for (int i = 0; i < 8; ++i) {
                const int row = rb + i + 8 * hi;
                const float v   = acc[i];
                const float pim = lpi[row * 8 + m];
                const float pv  = pim * v;
                s1[(size_t)row * S1W + c] += pv;   // distinct (row,c) per lane
                t1r[ib + i] += pv * v;
                if (m == lix[row]) outScores[(rowBase + row) * CLS + c] = v;
            }
        }
    };
    auto sg_ep = [&](int t, int rb, int ib, v8f acc) {
        const int j = (t - 27) * 16 + n;
        if (j < MIX * CLS) {
            const int m = j / CLS;
            #pragma unroll
            for (int i = 0; i < 8; ++i) {
                const int row = rb + i + 8 * hi;
                const float v  = acc[i];
                const float sp = (v > 20.0f) ? v : log1pf(__expf(v));
                alr[ib + i] += lpi[row * 8 + m] * sp;
            }
        }
    };
    auto bx_ep = [&](int t, int rb, v8f acc) {
        const int j = (t - 53) * 16 + n;
        if (j < BOXW) {
            #pragma unroll
            for (int i = 0; i < 8; ++i) {
                const int row = rb + i + 8 * hi;
                outDeltas[(rowBase + row) * BOXW + j] = acc[i];
            }
        }
    };

    // -------- mu tiles (13 passes of 2 tiles x 2 row tiles) ----------------
    for (int t = 1; t < 27; t += 2) {
        v8f a00, a01, a10, a11;
        gemm_quad(t, a00, a01, a10, a11);
        mu_ep(t, rb0, 0, a00); mu_ep(t + 1, rb0, 0, a01);
        mu_ep(t, rb1, 8, a10); mu_ep(t + 1, rb1, 8, a11);
    }
    // -------- sigma tiles --------------------------------------------------
    for (int t = 27; t < 53; t += 2) {
        v8f a00, a01, a10, a11;
        gemm_quad(t, a00, a01, a10, a11);
        sg_ep(t, rb0, 0, a00); sg_ep(t + 1, rb0, 0, a01);
        sg_ep(t, rb1, 8, a10); sg_ep(t + 1, rb1, 8, a11);
    }
    // -------- box tiles ----------------------------------------------------
    for (int t = 53; t < NTILES; t += 2) {
        v8f a00, a01, a10, a11;
        gemm_quad(t, a00, a01, a10, a11);
        bx_ep(t, rb0, a00); bx_ep(t + 1, rb0, a01);
        bx_ep(t, rb1, a10); bx_ep(t + 1, rb1, a11);
    }

    // -------- fold per-lane accumulators into per-row LDS scalars -----------
    #pragma unroll
    for (int q = 0; q < 2; ++q) {
        #pragma unroll
        for (int i = 0; i < 8; ++i) {
            const int row = 32 * w + 16 * q + i + 8 * hi;
            atomicAdd(&t1a[row], t1r[8 * q + i]);
            atomicAdd(&ala[row], alr[8 * q + i]);
        }
    }
    __threadfence_block();   // wave-local LDS is in-order; fence for safety

    // -------- per-row finish: epis = mean_c(sum pi mu^2 - (sum pi mu)^2) ----
    {
        const int row = rb0 + lane;          // all 32 lanes, one row each
        const size_t rg = rowBase + row;
        float ss = 0.0f;
        for (int c = 0; c < CLS; ++c) {
            const float s = s1[(size_t)row * S1W + c];
            ss += s * s;
        }
        outEpis[rg] = (t1a[row] - ss) * (1.0f / (float)CLS);
        outAlea[rg] = ala[row] * (1.0f / (float)CLS);
    }
    #undef LOAD_A
}

// ---------------------------------------------------------------------------
extern "C" void kernel_launch(void* const* d_in, const int* in_sizes, int n_in,
                              void* d_out, int out_size, void* d_ws, size_t ws_size,
                              hipStream_t stream)
{
    const float* x   = (const float*)d_in[0];
    const float* Wpi = (const float*)d_in[1];
    const float* bpi = (const float*)d_in[2];
    const float* Wmu = (const float*)d_in[3];
    const float* bmu = (const float*)d_in[4];
    const float* Wsg = (const float*)d_in[5];
    const float* bsg = (const float*)d_in[6];
    const float* Wbx = (const float*)d_in[7];
    const float* bbx = (const float*)d_in[8];
    float* out = (float*)d_out;

    const int N = in_sizes[0] / DIMK;

    __bf16* Bp    = (__bf16*)d_ws;
    float*  BiasP = (float*)((char*)d_ws + (size_t)NTILES * KCH * 32 * 16 * 2);

    // pack weights+biases (2.4 MB, re-done each call: deterministic)
    mln_pack<<<(NTILES * KCH * 32 + 255) / 256, 256, 0, stream>>>(
        Wpi, bpi, Wmu, bmu, Wsg, bsg, Wbx, bbx, Bp, BiasP);

    // fused GEMM + epilogue: 128 rows per WG, 4 waves, 313 KB dynamic LDS
    mln_fused<<<N / ROWS, WGT, SMEM_SZ, stream>>>(x, Bp, BiasP, out, N);
}
